// TriangleMultiplicationOutgoing_18373870092371
// MI455X (gfx1250) — matrix operand order repaired and checked by reading, not compiled, not run on hardware
//
#include <hip/hip_runtime.h>

// ---------------------------------------------------------------------------
// TriangleMultiplicationOutgoing for MI455X (gfx1250), bf16 WMMA pipeline.
// ---------------------------------------------------------------------------

#define NN   384          // sequence dim
#define DD   128          // d_model
#define CC   128          // channel
#define NCOL 640          // 4*CC (proj) + DD (gate)
#define RESW 644          // NCOL + 4 pad (bank-conflict avoidance)
#define MTOT (NN * NN)    // 147456 rows

typedef __attribute__((ext_vector_type(16))) __bf16 v16bf;
typedef __attribute__((ext_vector_type(8)))  float  v8f;

union Frag16 {
    v16bf bf;
    unsigned short u[16];
    uint4 q[2];
};

static __device__ __forceinline__ unsigned short f2bf(float f) {
    union { float f; unsigned u; } v; v.f = f;
    unsigned r = v.u + 0x7FFFu + ((v.u >> 16) & 1u);   // RNE
    return (unsigned short)(r >> 16);
}
static __device__ __forceinline__ float bf2f(unsigned short h) {
    union { unsigned u; float f; } v; v.u = ((unsigned)h) << 16;
    return v.f;
}
static __device__ __forceinline__ float sigmoidf(float x) {
    return 1.0f / (1.0f + __expf(-x));
}
static __device__ __forceinline__ v8f wmma_bf16(v16bf a, v16bf b, v8f c) {
    return __builtin_amdgcn_wmma_f32_16x16x32_bf16(
        /*neg_a=*/false, a, /*neg_b=*/false, b,
        /*c_mod=*/(short)0, c, /*reuse_a=*/false, /*reuse_b=*/false);
}

// ---------------------------------------------------------------------------
// k0: pack weights (f32 -> bf16) into WMMA B-fragment order.
// Layout: [ntile][kstep][lane(32)][elem(16)], B elem: K=(lane<16?0:16)+e,
// N = lane&15 within the 16-wide N tile.
// Wpk: 40 ntiles x 4 ksteps (cols 0..511 = W_proj, 512..639 = W_gate)
// Wabpk: 8 ntiles x 4 ksteps from W_ab.
// ---------------------------------------------------------------------------
__global__ __launch_bounds__(256) void k0_pack(
    const float* __restrict__ Wproj, const float* __restrict__ Wgate,
    const float* __restrict__ Wab,
    unsigned short* __restrict__ Wpk, unsigned short* __restrict__ Wabpk)
{
    int idx = blockIdx.x * 256 + threadIdx.x;   // 0 .. 81919
    {
        int e  = idx & 15;
        int ln = (idx >> 4) & 31;
        int kk = (idx >> 9) & 3;
        int nt = idx >> 11;                      // 0..39
        int n  = nt * 16 + (ln & 15);
        int k  = kk * 32 + ((ln < 16) ? 0 : 16) + e;
        float v = (n < 4 * CC) ? Wproj[k * (4 * CC) + n]
                               : Wgate[k * DD + (n - 4 * CC)];
        Wpk[idx] = f2bf(v);
    }
    if (idx < 8 * 4 * 32 * 16) {
        int e  = idx & 15;
        int ln = (idx >> 4) & 31;
        int kk = (idx >> 9) & 3;
        int nt = idx >> 11;                      // 0..7
        int n  = nt * 16 + (ln & 15);
        int k  = kk * 32 + ((ln < 16) ? 0 : 16) + e;
        Wabpk[idx] = f2bf(Wab[k * DD + n]);
    }
}

// ---------------------------------------------------------------------------
// k1: fused LayerNorm + [proj|gate] GEMM + sigmoid gating.
// Block = 16 rows (fixed i, 16 consecutive k), 256 threads = 8 waves.
// Outputs: aT,bT bf16 channel-major [c][i*NN+k]; g bf16 row-major [m][d].
// ---------------------------------------------------------------------------
__global__ __launch_bounds__(256) void k1_ln_proj(
    const float* __restrict__ x,
    const float* __restrict__ ln1_g, const float* __restrict__ ln1_b,
    const unsigned short* __restrict__ Wpk,
    unsigned short* __restrict__ aT, unsigned short* __restrict__ bT,
    unsigned short* __restrict__ gbuf)
{
    __shared__ float          xs[16][DD];     // 8 KB
    __shared__ unsigned short an[16][DD];     // 4 KB bf16 normalized
    __shared__ float          res[16][RESW];  // ~40 KB raw GEMM results

    const int t    = threadIdx.x;
    const int lane = t & 31;
    const int wave = t >> 5;
    const long m0  = (long)blockIdx.x * 16;

    // ---- stage x tile ----
    const float* xp = x + m0 * DD;
    #pragma unroll
    for (int it = 0; it < 8; ++it) {
        int idx = it * 256 + t;                // 0..2047
        xs[idx >> 7][idx & 127] = xp[idx];
    }
    __syncthreads();

    // ---- LayerNorm over d (row per half-wave, shfl_xor reduction) ----
    {
        int r = t >> 4, sub = t & 15;
        float s = 0.f, ss = 0.f;
        #pragma unroll
        for (int c = sub; c < DD; c += 16) { float v = xs[r][c]; s += v; ss += v * v; }
        #pragma unroll
        for (int off = 8; off >= 1; off >>= 1) {
            s  += __shfl_xor(s,  off, 16);
            ss += __shfl_xor(ss, off, 16);
        }
        float mu = s * (1.0f / DD);
        float rs = rsqrtf(ss * (1.0f / DD) - mu * mu + 1e-5f);
        #pragma unroll
        for (int c = sub; c < DD; c += 16) {
            float v = (xs[r][c] - mu) * rs * ln1_g[c] + ln1_b[c];
            an[r][c] = f2bf(v);
        }
    }
    __syncthreads();

    // ---- A fragments (contiguous k = 16h + s, same as Wpk), 4 K-steps ----
    Frag16 A[4];
    {
        int mrow = lane & 15;
        int kb   = (lane < 16) ? 0 : 16;
        #pragma unroll
        for (int kk = 0; kk < 4; ++kk) {
            A[kk].q[0] = *(const uint4*)&an[mrow][kk * 32 + kb];
            A[kk].q[1] = *(const uint4*)&an[mrow][kk * 32 + kb + 8];
        }
    }

    // ---- GEMM: 40 N-tiles, 5 per wave ----
    #pragma unroll
    for (int tt = 0; tt < 5; ++tt) {
        int nt = wave * 5 + tt;
        v8f acc = {};
        #pragma unroll
        for (int kk = 0; kk < 4; ++kk) {
            Frag16 B;
            const uint4* bp = (const uint4*)(Wpk + (size_t)((nt * 4 + kk) * 32 + lane) * 16);
            B.q[0] = bp[0]; B.q[1] = bp[1];
            acc = wmma_bf16(A[kk].bf, B.bf, acc);
        }
        int n0 = nt * 16 + (lane & 15);
        int mr = (lane < 16) ? 0 : 8;
        #pragma unroll
        for (int v = 0; v < 8; ++v) res[mr + v][n0] = acc[v];
    }
    __syncthreads();

    // ---- epilogue: sigmoid gating, scatter channel-major a/b + row-major g ----
    const int  i  = (int)(m0 / NN);
    const int  k0 = (int)(m0 % NN);
    const long base = (long)i * NN + k0;
    {
        int r = t & 15, chi = t >> 4;          // r varies fastest -> contig k
        #pragma unroll
        for (int cg = 0; cg < 8; ++cg) {
            int c = cg * 16 + chi;
            float av = sigmoidf(res[r][c])          * res[r][CC + c];
            float bv = sigmoidf(res[r][2 * CC + c]) * res[r][3 * CC + c];
            aT[(long)c * (NN * (long)NN) + base + r] = f2bf(av);
            bT[(long)c * (NN * (long)NN) + base + r] = f2bf(bv);
        }
    }
    {
        int rr = t >> 4, cc = t & 15;          // cc varies fastest -> contig c
        #pragma unroll
        for (int cg = 0; cg < 8; ++cg) {
            int c = cg * 16 + cc;
            gbuf[(m0 + rr) * (long)DD + c] = f2bf(sigmoidf(res[rr][4 * CC + c]));
        }
    }
}

// ---------------------------------------------------------------------------
// k2: triangle einsum, per channel: AB = A * B^T (both [row][k] bf16).
// Block: one channel, 64x64 output tile, 8 waves x 2 accumulators, K=384.
// Row-major [row][k] storage matches the WMMA B (K-major per lane) layout,
// so every fragment load is contiguous.
// ---------------------------------------------------------------------------
__global__ __launch_bounds__(256) void k2_triangle(
    const unsigned short* __restrict__ aT,
    const unsigned short* __restrict__ bT,
    float* __restrict__ abT)
{
    const int t    = threadIdx.x;
    const int lane = t & 31;
    const int wave = t >> 5;
    const int c    = blockIdx.z;
    const int i0   = blockIdx.y * 64;
    const int j0   = blockIdx.x * 64;

    const unsigned short* Ap = aT + (long)c * NN * NN;
    const unsigned short* Bp = bT + (long)c * NN * NN;

    const int mt  = wave & 3;
    const int ntb = (wave >> 2) * 2;

    const int mrow  = i0 + mt * 16 + (lane & 15);
    const int kbA   = (lane < 16) ? 0 : 16;
    const int jrow0 = j0 + ntb * 16 + (lane & 15);
    const int jrow1 = jrow0 + 16;
    const int kbB   = (lane < 16) ? 0 : 16;

    const unsigned short* ap  = Ap + (long)mrow  * NN + kbA;
    const unsigned short* bp0 = Bp + (long)jrow0 * NN + kbB;
    const unsigned short* bp1 = Bp + (long)jrow1 * NN + kbB;

    v8f acc0 = {}, acc1 = {};
    for (int kt = 0; kt < NN; kt += 32) {
        Frag16 A, B0, B1;
        A.q[0]  = *(const uint4*)(ap + kt);
        A.q[1]  = *(const uint4*)(ap + kt + 8);
        B0.q[0] = *(const uint4*)(bp0 + kt);
        B0.q[1] = *(const uint4*)(bp0 + kt + 8);
        B1.q[0] = *(const uint4*)(bp1 + kt);
        B1.q[1] = *(const uint4*)(bp1 + kt + 8);
        if (kt + 32 < NN) {                    // emits global_prefetch_b8
            __builtin_prefetch((const void*)(ap  + kt + 32), 0, 1);
            __builtin_prefetch((const void*)(bp0 + kt + 32), 0, 1);
            __builtin_prefetch((const void*)(bp1 + kt + 32), 0, 1);
        }
        acc0 = wmma_bf16(A.bf, B0.bf, acc0);
        acc1 = wmma_bf16(A.bf, B1.bf, acc1);
    }

    float* outp = abT + (long)c * NN * NN;
    const int Mb  = i0 + mt * 16 + ((lane < 16) ? 0 : 8);
    const int nj0 = j0 + ntb * 16 + (lane & 15);
    #pragma unroll
    for (int v = 0; v < 8; ++v) {
        outp[(long)(Mb + v) * NN + nj0]      = acc0[v];
        outp[(long)(Mb + v) * NN + nj0 + 16] = acc1[v];
    }
}

// ---------------------------------------------------------------------------
// k3: LayerNorm over c + (@ W_ab) GEMM + multiply by gate g.
// Block = 16 rows (fixed i, 16 consecutive j), 8 waves x 1 N-tile of d.
// ---------------------------------------------------------------------------
__global__ __launch_bounds__(256) void k3_out(
    const float* __restrict__ abT,
    const float* __restrict__ ln2_g, const float* __restrict__ ln2_b,
    const unsigned short* __restrict__ Wabpk,
    const unsigned short* __restrict__ gbuf,
    float* __restrict__ out)
{
    __shared__ float          abs_[16][CC + 4];  // [j][c], padded
    __shared__ unsigned short an[16][CC];        // normalized bf16
    __shared__ unsigned short gs[16][DD];        // gate tile

    const int t    = threadIdx.x;
    const int lane = t & 31;
    const int wave = t >> 5;
    const long m0  = (long)blockIdx.x * 16;
    const int  i   = (int)(m0 / NN);
    const int  j0  = (int)(m0 % NN);

    // ---- stage ab tile (channel-major source) ----
    {
        int jj = t & 15, chi = t >> 4;
        #pragma unroll
        for (int cg = 0; cg < 8; ++cg) {
            int c = cg * 16 + chi;
            abs_[jj][c] = abT[(long)c * NN * NN + (long)i * NN + j0 + jj];
        }
    }
    // ---- stage gate tile ----
    {
        int r = t >> 4, c0 = (t & 15) * 8;
        *(uint4*)&gs[r][c0] = *(const uint4*)(gbuf + (m0 + r) * (long)DD + c0);
    }
    __syncthreads();

    // ---- LayerNorm over c ----
    {
        int r = t >> 4, sub = t & 15;
        float s = 0.f, ss = 0.f;
        #pragma unroll
        for (int c = sub; c < CC; c += 16) { float v = abs_[r][c]; s += v; ss += v * v; }
        #pragma unroll
        for (int off = 8; off >= 1; off >>= 1) {
            s  += __shfl_xor(s,  off, 16);
            ss += __shfl_xor(ss, off, 16);
        }
        float mu = s * (1.0f / CC);
        float rs = rsqrtf(ss * (1.0f / CC) - mu * mu + 1e-5f);
        #pragma unroll
        for (int c = sub; c < CC; c += 16) {
            float v = (abs_[r][c] - mu) * rs * ln2_g[c] + ln2_b[c];
            an[r][c] = f2bf(v);
        }
    }
    __syncthreads();

    // ---- GEMM [16 x 128] x [128 x 128], wave -> one 16-wide d tile ----
    v8f acc = {};
    {
        int mrow = lane & 15;
        int kb   = (lane < 16) ? 0 : 16;
        #pragma unroll
        for (int kk = 0; kk < 4; ++kk) {
            Frag16 Af, B;
            Af.q[0] = *(const uint4*)&an[mrow][kk * 32 + kb];
            Af.q[1] = *(const uint4*)&an[mrow][kk * 32 + kb + 8];
            const uint4* bp = (const uint4*)(Wabpk + (size_t)((wave * 4 + kk) * 32 + lane) * 16);
            B.q[0] = bp[0]; B.q[1] = bp[1];
            acc = wmma_bf16(Af.bf, B.bf, acc);
        }
    }

    // ---- epilogue: out = g * (LN(ab) @ W_ab) ----
    const int Mb = (lane < 16) ? 0 : 8;
    const int nn = wave * 16 + (lane & 15);
    #pragma unroll
    for (int v = 0; v < 8; ++v) {
        float gv = bf2f(gs[Mb + v][nn]);
        out[(m0 + Mb + v) * (long)DD + nn] = acc[v] * gv;
    }
}

// ---------------------------------------------------------------------------
// Launch
// ---------------------------------------------------------------------------
extern "C" void kernel_launch(void* const* d_in, const int* in_sizes, int n_in,
                              void* d_out, int out_size, void* d_ws, size_t ws_size,
                              hipStream_t stream) {
    (void)in_sizes; (void)n_in; (void)out_size; (void)ws_size;
    const float* x      = (const float*)d_in[0];
    const float* ln1_g  = (const float*)d_in[1];
    const float* ln1_b  = (const float*)d_in[2];
    const float* W_proj = (const float*)d_in[3];
    const float* W_gate = (const float*)d_in[4];
    const float* ln2_g  = (const float*)d_in[5];
    const float* ln2_b  = (const float*)d_in[6];
    const float* W_ab   = (const float*)d_in[7];
    float* out = (float*)d_out;

    char* ws = (char*)d_ws;
    const size_t SZ_WPK   = 40 * 4 * 32 * 16 * sizeof(unsigned short); // 163840
    const size_t SZ_WABPK = 8 * 4 * 32 * 16 * sizeof(unsigned short);  // 32768
    const size_t SZ_PLANE = (size_t)CC * NN * NN;                      // 18874368 elems

    unsigned short* Wpk   = (unsigned short*)ws;
    unsigned short* Wabpk = (unsigned short*)(ws + SZ_WPK);
    unsigned short* aT    = (unsigned short*)(ws + SZ_WPK + SZ_WABPK);
    unsigned short* bT    = aT + SZ_PLANE;
    unsigned short* gbuf  = bT + SZ_PLANE;
    float*          abT   = (float*)(ws + SZ_WPK + SZ_WABPK + 3 * SZ_PLANE * sizeof(unsigned short));

    k0_pack<<<320, 256, 0, stream>>>(W_proj, W_gate, W_ab, Wpk, Wabpk);
    k1_ln_proj<<<MTOT / 16, 256, 0, stream>>>(x, ln1_g, ln1_b, Wpk, aT, bT, gbuf);
    k2_triangle<<<dim3(NN / 64, NN / 64, CC), 256, 0, stream>>>(aT, bT, abT);
    k3_out<<<MTOT / 16, 256, 0, stream>>>(abT, ln2_g, ln2_b, Wabpk, gbuf, out);
}
